// BlockWithBias_56229711839984
// MI455X (gfx1250) — matrix operand order repaired and checked by reading, not compiled, not run on hardware
//
#include <hip/hip_runtime.h>
#include <hip/hip_bf16.h>
#include <stdint.h>

typedef __attribute__((ext_vector_type(16))) __bf16 v16bf;
typedef __attribute__((ext_vector_type(8)))  float  v8f;
typedef __attribute__((ext_vector_type(4)))  unsigned int u32x4;
typedef __attribute__((ext_vector_type(8)))  int i32x8;
typedef __attribute__((ext_vector_type(4)))  int i32x4;

constexpr int SEQ    = 2048;
constexpr int HIDN   = 1024;
constexpr int NHEAD  = 16;
constexpr int HDIM   = 64;
constexpr int INTERM = 2816;

union FragBF { v16bf v; uint16_t u[16]; uint32_t w[8]; };
union Acc8   { v8f v; float f[8]; };

__device__ __forceinline__ uint16_t f2bf(float x) {
  uint32_t u = __float_as_uint(x);
  u += 0x7FFFu + ((u >> 16) & 1u);   // round-to-nearest-even
  return (uint16_t)(u >> 16);
}

__device__ __forceinline__ v8f wmma_bf16(v16bf a, v16bf b, v8f c) {
  // D = A(16x32 bf16) * B(32x16 bf16) + C(16x16 f32)
  return __builtin_amdgcn_wmma_f32_16x16x32_bf16(false, a, false, b, (short)0, c,
                                                 false, false);
}

// interleave two rows of 8 halves into 8 (lo,hi) 32-bit pairs via v_perm_b32
__device__ __forceinline__ void interleave8(const uint4& r0, const uint4& r1,
                                            uint32_t* w) {
  const uint32_t* a = (const uint32_t*)&r0;
  const uint32_t* b = (const uint32_t*)&r1;
#pragma unroll
  for (int j = 0; j < 4; ++j) {
    w[2 * j]     = __builtin_amdgcn_perm(b[j], a[j], 0x05040100u);
    w[2 * j + 1] = __builtin_amdgcn_perm(b[j], a[j], 0x07060302u);
  }
}

// ---------------------------------------------------------------------------
// TDM: async 2D bf16 tile load Global -> LDS (tensor_load_to_lds).
// D# per CDNA5 ISA 8.3/8.4: group0 = {count=1, lds_addr, global_addr, type=2},
// group1 = {data_size=2B, tensor_dim0/1, tile_dim0/1, dim0_stride}.
// Issue from one wave; completion via s_wait_tensorcnt. EXEC ignored by TDM.
// ---------------------------------------------------------------------------
__device__ __forceinline__ void tdm_load_bf16_2d(
    uint32_t lds_byte_off, const uint16_t* gsrc, uint32_t tile_x, uint32_t tile_y,
    uint32_t tensor_x, uint32_t tensor_y, uint64_t row_stride_elems) {
  const uint64_t ga = (uint64_t)(uintptr_t)gsrc;
  const u32x4 g0 = {
      1u,                                                  // count=1 (valid)
      lds_byte_off,                                        // lds_addr
      (uint32_t)ga,                                        // global_addr[31:0]
      (uint32_t)((ga >> 32) & 0x01FFFFFFu) | (2u << 30)};  // addr[56:32], type=2
  const i32x8 g1 = {
      (int)(1u << 16),                                     // data_size=1 (2B)
      (int)((tensor_x & 0xFFFFu) << 16),                   // tensor_dim0[15:0]
      (int)((tensor_x >> 16) | ((tensor_y & 0xFFFFu) << 16)),
      (int)((tensor_y >> 16) | (tile_x << 16)),            // tile_dim0
      (int)tile_y,                                         // tile_dim1 (dim2=0)
      (int)(uint32_t)row_stride_elems,                     // dim0_stride[31:0]
      (int)(uint32_t)((row_stride_elems >> 32) & 0xFFFFu), // dim0_stride[47:32]
      0};
  const i32x4 gz = {0, 0, 0, 0};
#if defined(__clang_major__) && __clang_major__ >= 23
  const i32x8 gz8 = {0, 0, 0, 0, 0, 0, 0, 0};
  __builtin_amdgcn_tensor_load_to_lds(g0, g1, gz, gz, gz8, 0);
#else
  __builtin_amdgcn_tensor_load_to_lds(g0, g1, gz, gz, 0);
#endif
}

// ---------------------------------------------------------------------------
// f32 -> bf16 conversion
// ---------------------------------------------------------------------------
__global__ __launch_bounds__(256) void cvt_f32_bf16_kernel(
    const float* __restrict__ in, uint16_t* __restrict__ out, size_t n) {
  size_t i = (size_t)blockIdx.x * 256 + threadIdx.x;
  if (i < n) out[i] = f2bf(in[i]);
}

// ---------------------------------------------------------------------------
// Tiled bf16 GEMM: C[M,N] f32 = A[M,K] bf16 (row-major) * B[K,N] bf16 (row-major)
// Block: 256 threads (8 waves); block tile 64x128, K stepped by 32.
// Each wave owns a 32x32 C tile: 2 A-frags x 2 B-frags -> 4 WMMAs per K-step.
// A tiles are DMA'd into LDS by the Tensor Data Mover (wave 0 issues, waits
// TENSORcnt before the iteration barrier); B tiles staged manually with
// k-pair interleave (v_perm_b32) so B-fragments are 8 conflict-free
// ds_load_b32. Double-buffered: one barrier per K-step.
// Dynamic LDS layout: A bufs @ [0,8192), B bufs @ [8192,24576).
// M%64==0, N%128==0, K%32==0 (true for all call sites).
// ---------------------------------------------------------------------------
__global__ __launch_bounds__(256) void gemm_bf16_kernel(
    const uint16_t* __restrict__ A, const uint16_t* __restrict__ B,
    float* __restrict__ C, int M, int N, int K) {
  extern __shared__ __align__(16) char smem[];
  uint16_t (*As)[64][32]  = (uint16_t(*)[64][32])smem;            // 2 x 4096 B
  uint32_t (*Bs)[16][128] = (uint32_t(*)[16][128])(smem + 8192);  // 2 x 8192 B
  const int tid  = threadIdx.x;
  const int lane = tid & 31;
  const int wave = tid >> 5;
  const int wm   = (wave & 1) * 32;   // wave row in block tile (2 m-waves)
  const int wn   = (wave >> 1) * 32;  // wave col in block tile (4 n-waves)
  const int bm   = blockIdx.x * 64;
  const int bn   = blockIdx.y * 128;

  Acc8 acc[2][2];
#pragma unroll
  for (int i = 0; i < 2; ++i)
#pragma unroll
    for (int j = 0; j < 2; ++j)
#pragma unroll
      for (int e = 0; e < 8; ++e) acc[i][j].f[e] = 0.f;

  const int bkp = tid >> 4, bnb = (tid & 15) * 8;   // B stage: pair-row, 8 cols
  const int row = lane & 15;
  const int kha = (lane >> 4) * 16;  // A-frag K base per half-wave
  const int khp = (lane >> 4) * 8;   // B-frag k-pair base per half-wave

  const uint16_t* atile = A + (size_t)bm * K;                      // tile origin
  const uint16_t* bptr0 = B + (size_t)(2 * bkp) * N + bn + bnb;    // even k row
  const uint16_t* bptr1 = bptr0 + N;                               // odd k row

  auto stageA = [&](int buf, int k0) {   // async TDM tile load, wave 0 only
    if (wave == 0)
      tdm_load_bf16_2d((uint32_t)buf * 4096u, atile + k0, /*tile_x=*/32,
                       /*tile_y=*/64, /*tensor_x=*/(uint32_t)K,
                       /*tensor_y=*/(uint32_t)M, /*stride=*/(uint64_t)K);
  };
  auto stageB = [&](int buf, int k0) {
    const uint4 r0 = *(const uint4*)(bptr0 + (size_t)k0 * N);
    const uint4 r1 = *(const uint4*)(bptr1 + (size_t)k0 * N);
    uint32_t w[8];
    interleave8(r0, r1, w);
    *(uint4*)&Bs[buf][bkp][bnb]     = *(uint4*)&w[0];
    *(uint4*)&Bs[buf][bkp][bnb + 4] = *(uint4*)&w[4];
  };

  const int nk = K >> 5;
  stageA(0, 0);
  stageB(0, 0);
  for (int kt = 0; kt < nk; ++kt) {
    if (wave == 0) __builtin_amdgcn_s_wait_tensorcnt(0);  // tile kt in LDS
    __syncthreads();
    if (kt + 1 < nk) {
      stageA((kt + 1) & 1, (kt + 1) * 32);
      stageB((kt + 1) & 1, (kt + 1) * 32);
    }
    if (kt + 2 < nk)   // global_prefetch_b8 of B two tiles ahead
      __builtin_prefetch(bptr0 + (size_t)(kt + 2) * 32 * N, 0, 3);
    const int cb = kt & 1;

    FragBF a[2], b[2];
#pragma unroll
    for (int t = 0; t < 2; ++t) {   // A 16x32 layout: VGPR j -> contiguous K pair
      const uint16_t* arow = &As[cb][wm + t * 16 + row][0];
#pragma unroll
      for (int j = 0; j < 4; ++j) {
        a[t].w[j]     = *(const uint32_t*)&arow[kha + 2 * j];
        a[t].w[4 + j] = *(const uint32_t*)&arow[kha + 8 + 2 * j];
      }
    }
#pragma unroll
    for (int t = 0; t < 2; ++t) {   // B frag: 8 ds_load_b32, pre-interleaved
      const int col = wn + t * 16 + row;
#pragma unroll
      for (int j = 0; j < 8; ++j) b[t].w[j] = Bs[cb][khp + j][col];
    }
#pragma unroll
    for (int i = 0; i < 2; ++i)
#pragma unroll
      for (int j = 0; j < 2; ++j)
        acc[i][j].v = wmma_bf16(a[i].v, b[j].v, acc[i][j].v);
  }

#pragma unroll
  for (int i = 0; i < 2; ++i) {
    const int crow = bm + wm + i * 16 + (lane >> 4) * 8;  // C row = vgpr+8*half
#pragma unroll
    for (int j = 0; j < 2; ++j) {
      const int ccol = bn + wn + j * 16 + row;
#pragma unroll
      for (int e = 0; e < 8; ++e)
        C[(size_t)(crow + e) * N + ccol] = acc[i][j].f[e];
    }
  }
}

// ---------------------------------------------------------------------------
// RoPE: qkv f32 [S][3*NH][HD] -> q bf16 [NH][S][HD], kT bf16 [NH][HD][S],
//       v bf16 [NH][S][HD]
// ---------------------------------------------------------------------------
__global__ __launch_bounds__(256) void rope_kernel(
    const float* __restrict__ qkv, const float* __restrict__ cosp,
    const float* __restrict__ sinp, uint16_t* __restrict__ qb,
    uint16_t* __restrict__ ktb, uint16_t* __restrict__ vb) {
  size_t t = (size_t)blockIdx.x * 256 + threadIdx.x;
  if (t >= (size_t)SEQ * NHEAD * HDIM) return;
  const int d = (int)(t & 63);
  const int h = (int)((t >> 6) & 15);
  const int s = (int)(t >> 10);
  const float* base = qkv + (size_t)s * (3 * NHEAD * HDIM);
  const float cs = cosp[s * HDIM + d];
  const float sn = sinp[s * HDIM + d];
  const float qv = base[h * HDIM + d];
  const float qp = base[h * HDIM + (d ^ 32)];
  const float kv = base[(NHEAD + h) * HDIM + d];
  const float kp = base[(NHEAD + h) * HDIM + (d ^ 32)];
  const float vv = base[(2 * NHEAD + h) * HDIM + d];
  const float qr = (d < 32) ? (qv * cs - qp * sn) : (qv * cs + qp * sn);
  const float kr = (d < 32) ? (kv * cs - kp * sn) : (kv * cs + kp * sn);
  const size_t hs = (size_t)h * SEQ + s;
  qb[hs * HDIM + d] = f2bf(qr);
  vb[hs * HDIM + d] = f2bf(vv);
  ktb[((size_t)h * HDIM + d) * SEQ + s] = f2bf(kr);
}

// ---------------------------------------------------------------------------
// Flash attention (FA2-style): one wave per (16-row Q tile, head).
// scores = (Q·K^T)*scale + bias ; online softmax ; O += P·V ; O /= l
// K^T and V tiles staged into LDS k-pair-interleaved so all B-fragment builds
// are ds_load_b32. Streams the 256MB attn_bias exactly once (dominant traffic).
// ---------------------------------------------------------------------------
__global__ __launch_bounds__(32) void flash_attn_kernel(
    const uint16_t* __restrict__ q, const uint16_t* __restrict__ kt,
    const uint16_t* __restrict__ v, const float* __restrict__ bias,
    float* __restrict__ out) {
  __shared__ __align__(16) uint32_t Ktp[32][32];  // [dp][s]: halves (2dp,2dp+1)
  __shared__ __align__(16) uint32_t Vp[16][64];   // [sp][d]: halves (2sp,2sp+1)
  __shared__ __align__(16) uint16_t Ps[16][32];   // P tile bf16
  const int lane = threadIdx.x;
  const int qb   = blockIdx.x * 16;
  const int h    = blockIdx.y;
  const int row  = lane & 15;
  const int half = lane >> 4;
  const int kha  = half * 16;
  const int khp  = half * 8;
  const float scale = 0.125f;  // HD^-0.5

  FragBF qa[2];  // Q tile 16x64 as two 16x32 A fragments (held in regs)
  {
    const uint16_t* qrow = q + ((size_t)h * SEQ + qb + row) * HDIM;
#pragma unroll
    for (int c = 0; c < 2; ++c) {
      const int db = c * 32;
#pragma unroll
      for (int j = 0; j < 4; ++j) {
        qa[c].w[j]     = *(const uint32_t*)&qrow[db + kha + 2 * j];
        qa[c].w[4 + j] = *(const uint32_t*)&qrow[db + kha + 8 + 2 * j];
      }
    }
  }

  Acc8 o[4];
  float m[8], l[8];
#pragma unroll
  for (int i = 0; i < 8; ++i) { m[i] = -3.4e38f; l[i] = 0.f; }
#pragma unroll
  for (int c = 0; c < 4; ++c)
#pragma unroll
    for (int i = 0; i < 8; ++i) o[c].f[i] = 0.f;

  for (int kb = 0; kb < SEQ; kb += 32) {
    // stage K^T tile (pairs over d) and V tile (pairs over s) into LDS
    {
      const uint16_t* r0 = kt + ((size_t)h * HDIM + 2 * lane) * SEQ + kb;
      const uint16_t* r1 = r0 + SEQ;
#pragma unroll
      for (int qd = 0; qd < 4; ++qd) {
        uint32_t w[8];
        interleave8(*(const uint4*)(r0 + qd * 8), *(const uint4*)(r1 + qd * 8), w);
        *(uint4*)&Ktp[lane][qd * 8]     = *(uint4*)&w[0];
        *(uint4*)&Ktp[lane][qd * 8 + 4] = *(uint4*)&w[4];
      }
    }
    {
      const int sp = lane >> 1, db = (lane & 1) * 32;
      const uint16_t* r0 = v + ((size_t)h * SEQ + kb + 2 * sp) * HDIM + db;
      const uint16_t* r1 = r0 + HDIM;
#pragma unroll
      for (int qd = 0; qd < 4; ++qd) {
        uint32_t w[8];
        interleave8(*(const uint4*)(r0 + qd * 8), *(const uint4*)(r1 + qd * 8), w);
        *(uint4*)&Vp[sp][db + qd * 8]     = *(uint4*)&w[0];
        *(uint4*)&Vp[sp][db + qd * 8 + 4] = *(uint4*)&w[4];
      }
    }
    __syncthreads();

    // scores: two 16x16 tiles covering this 32-key block
    Acc8 s[2];
#pragma unroll
    for (int nc = 0; nc < 2; ++nc) {
#pragma unroll
      for (int i = 0; i < 8; ++i) s[nc].f[i] = 0.f;
      const int n = nc * 16 + row;
#pragma unroll
      for (int dc = 0; dc < 2; ++dc) {
        FragBF bf;
#pragma unroll
        for (int j = 0; j < 8; ++j) bf.w[j] = Ktp[dc * 16 + khp + j][n];
        s[nc].v = wmma_bf16(qa[dc].v, bf.v, s[nc].v);
      }
      const float* bp = bias + ((size_t)h * SEQ + qb + half * 8) * SEQ + kb +
                        nc * 16 + row;
      __builtin_prefetch(bp + 32, 0, 3);   // warm next key-block of the stream
#pragma unroll
      for (int i = 0; i < 8; ++i)
        s[nc].f[i] = s[nc].f[i] * scale + bp[(size_t)i * SEQ];
    }

    // online softmax: row max/sum via 16-lane xor shuffles
    float rmax[8], rsum[8], alpha[8];
#pragma unroll
    for (int i = 0; i < 8; ++i) rmax[i] = fmaxf(s[0].f[i], s[1].f[i]);
#pragma unroll
    for (int mk = 1; mk < 16; mk <<= 1)
#pragma unroll
      for (int i = 0; i < 8; ++i)
        rmax[i] = fmaxf(rmax[i], __shfl_xor(rmax[i], mk, 32));
#pragma unroll
    for (int i = 0; i < 8; ++i) {
      const float nm = fmaxf(m[i], rmax[i]);
      alpha[i] = __expf(m[i] - nm);
      m[i] = nm;
      s[0].f[i] = __expf(s[0].f[i] - nm);
      s[1].f[i] = __expf(s[1].f[i] - nm);
      rsum[i] = s[0].f[i] + s[1].f[i];
    }
#pragma unroll
    for (int mk = 1; mk < 16; mk <<= 1)
#pragma unroll
      for (int i = 0; i < 8; ++i) rsum[i] += __shfl_xor(rsum[i], mk, 32);
#pragma unroll
    for (int i = 0; i < 8; ++i) l[i] = l[i] * alpha[i] + rsum[i];
#pragma unroll
    for (int c = 0; c < 4; ++c)
#pragma unroll
      for (int i = 0; i < 8; ++i) o[c].f[i] *= alpha[i];

    // P (C-layout) -> LDS bf16 -> A-fragment layout
#pragma unroll
    for (int i = 0; i < 8; ++i) {
      Ps[half * 8 + i][row]      = f2bf(s[0].f[i]);
      Ps[half * 8 + i][16 + row] = f2bf(s[1].f[i]);
    }
    __syncthreads();

    FragBF pa;
#pragma unroll
    for (int j = 0; j < 4; ++j) {
      pa.w[j]     = *(const uint32_t*)&Ps[row][kha + 2 * j];
      pa.w[4 + j] = *(const uint32_t*)&Ps[row][kha + 8 + 2 * j];
    }
#pragma unroll
    for (int c = 0; c < 4; ++c) {
      FragBF vb;
      const int n = c * 16 + row;
#pragma unroll
      for (int j = 0; j < 8; ++j) vb.w[j] = Vp[khp + j][n];
      o[c].v = wmma_bf16(pa.v, vb.v, o[c].v);
    }
    __syncthreads();
  }

#pragma unroll
  for (int i = 0; i < 8; ++i) {
    const float inv = 1.f / l[i];
    const size_t rbase =
        (size_t)(qb + half * 8 + i) * (NHEAD * HDIM) + (size_t)h * HDIM;
#pragma unroll
    for (int c = 0; c < 4; ++c) out[rbase + c * 16 + row] = o[c].f[i] * inv;
  }
}

// ---------------------------------------------------------------------------
// y = rmsnorm(a + b); writes f32 and (optionally) bf16. One block per row.
// ---------------------------------------------------------------------------
__global__ __launch_bounds__(256) void add_rmsnorm_kernel(
    const float* __restrict__ a, const float* __restrict__ b,
    float* __restrict__ outf, uint16_t* __restrict__ outb) {
  __shared__ float wsum[8];
  const int t = threadIdx.x;
  const size_t base = (size_t)blockIdx.x * HIDN;
  float xv[4];
  float ss = 0.f;
#pragma unroll
  for (int j = 0; j < 4; ++j) {
    const int c = t + j * 256;
    xv[j] = a[base + c] + b[base + c];
    ss += xv[j] * xv[j];
  }
#pragma unroll
  for (int mk = 16; mk >= 1; mk >>= 1) ss += __shfl_xor(ss, mk, 32);
  if ((t & 31) == 0) wsum[t >> 5] = ss;
  __syncthreads();
  float tot = 0.f;
#pragma unroll
  for (int w = 0; w < 8; ++w) tot += wsum[w];
  const float inv = rsqrtf(tot * (1.f / HIDN) + 1e-5f);
#pragma unroll
  for (int j = 0; j < 4; ++j) {
    const int c = t + j * 256;
    const float y = xv[j] * inv;
    outf[base + c] = y;
    if (outb) outb[base + c] = f2bf(y);
  }
}

// ---------------------------------------------------------------------------
// h = silu(gate) * up, bf16 out
// ---------------------------------------------------------------------------
__global__ __launch_bounds__(256) void swiglu_kernel(
    const float* __restrict__ gu, uint16_t* __restrict__ hb) {
  size_t t = (size_t)blockIdx.x * 256 + threadIdx.x;
  if (t >= (size_t)SEQ * INTERM) return;
  const size_t s = t / INTERM, c = t % INTERM;
  const float g = gu[s * (2 * INTERM) + c];
  const float u = gu[s * (2 * INTERM) + INTERM + c];
  const float sig = 1.f / (1.f + __expf(-g));
  hb[t] = f2bf(g * sig * u);
}

// ---------------------------------------------------------------------------
extern "C" void kernel_launch(void* const* d_in, const int* in_sizes, int n_in,
                              void* d_out, int out_size, void* d_ws,
                              size_t ws_size, hipStream_t stream) {
  (void)in_sizes; (void)n_in; (void)out_size; (void)ws_size;
  const float* cosp = (const float*)d_in[0];
  const float* sinp = (const float*)d_in[1];
  const float* x    = (const float*)d_in[2];
  const float* bias = (const float*)d_in[3];
  const float* wqkv = (const float*)d_in[4];
  const float* wo   = (const float*)d_in[5];
  const float* wgu  = (const float*)d_in[6];
  const float* wd   = (const float*)d_in[7];

  char* ws = (char*)d_ws;
  // long-lived region
  const size_t off_x1f = 0;                                     // f32 [S,H]
  const size_t off_x1b = off_x1f + (size_t)SEQ * HIDN * 4;      // bf16 [S,H]
  const size_t A0      = off_x1b + (size_t)SEQ * HIDN * 2;
  // phase 1: qkv projection + rope
  const size_t off_xb   = A0;
  const size_t off_wqkv = off_xb   + (size_t)SEQ * HIDN * 2;
  const size_t off_qkvf = off_wqkv + (size_t)HIDN * 3 * NHEAD * HDIM * 2;
  const size_t off_qb   = off_qkvf + (size_t)SEQ * 3 * NHEAD * HDIM * 4;
  const size_t off_ktb  = off_qb   + (size_t)NHEAD * SEQ * HDIM * 2;
  const size_t off_vb   = off_ktb  + (size_t)NHEAD * SEQ * HDIM * 2;
  // phase 2: attention out + o-proj (reuses phase-1 space, lifetimes disjoint)
  const size_t off_af  = A0;
  const size_t off_afb = off_af  + (size_t)SEQ * HIDN * 4;
  const size_t off_wob = off_afb + (size_t)SEQ * HIDN * 2;
  const size_t off_apf = off_wob + (size_t)HIDN * HIDN * 2;
  // phase 3: MLP (reuses phase-1/2 space)
  const size_t off_wgub = A0;
  const size_t off_guf  = off_wgub + (size_t)HIDN * 2 * INTERM * 2;
  const size_t off_hb   = off_guf  + (size_t)SEQ * 2 * INTERM * 4;
  const size_t off_wdb  = off_hb   + (size_t)SEQ * INTERM * 2;
  const size_t off_mlpf = off_wdb  + (size_t)INTERM * HIDN * 2;

  auto bfp = [&](size_t off) { return (uint16_t*)(ws + off); };
  auto ffp = [&](size_t off) { return (float*)(ws + off); };
  auto cvt = [&](const float* src, uint16_t* dst, size_t n) {
    cvt_f32_bf16_kernel<<<(unsigned)((n + 255) / 256), 256, 0, stream>>>(src, dst, n);
  };
  constexpr unsigned GEMM_LDS = 24576;   // 8KB A bufs + 16KB B bufs

  // ---- attention path ----
  cvt(x,    bfp(off_xb),   (size_t)SEQ * HIDN);
  cvt(wqkv, bfp(off_wqkv), (size_t)HIDN * 3 * NHEAD * HDIM);
  gemm_bf16_kernel<<<dim3(SEQ / 64, (3 * NHEAD * HDIM) / 128), 256, GEMM_LDS,
                    stream>>>(bfp(off_xb), bfp(off_wqkv), ffp(off_qkvf), SEQ,
                              3 * NHEAD * HDIM, HIDN);
  {
    size_t n = (size_t)SEQ * NHEAD * HDIM;
    rope_kernel<<<(unsigned)((n + 255) / 256), 256, 0, stream>>>(
        ffp(off_qkvf), cosp, sinp, bfp(off_qb), bfp(off_ktb), bfp(off_vb));
  }
  flash_attn_kernel<<<dim3(SEQ / 16, NHEAD), 32, 0, stream>>>(
      bfp(off_qb), bfp(off_ktb), bfp(off_vb), bias, ffp(off_af));
  cvt(ffp(off_af), bfp(off_afb), (size_t)SEQ * HIDN);
  cvt(wo, bfp(off_wob), (size_t)HIDN * HIDN);
  gemm_bf16_kernel<<<dim3(SEQ / 64, HIDN / 128), 256, GEMM_LDS, stream>>>(
      bfp(off_afb), bfp(off_wob), ffp(off_apf), SEQ, HIDN, HIDN);
  add_rmsnorm_kernel<<<SEQ, 256, 0, stream>>>(x, ffp(off_apf), ffp(off_x1f),
                                              bfp(off_x1b));

  // ---- MLP path ----
  cvt(wgu, bfp(off_wgub), (size_t)HIDN * 2 * INTERM);
  gemm_bf16_kernel<<<dim3(SEQ / 64, (2 * INTERM) / 128), 256, GEMM_LDS, stream>>>(
      bfp(off_x1b), bfp(off_wgub), ffp(off_guf), SEQ, 2 * INTERM, HIDN);
  {
    size_t n = (size_t)SEQ * INTERM;
    swiglu_kernel<<<(unsigned)((n + 255) / 256), 256, 0, stream>>>(
        ffp(off_guf), bfp(off_hb));
  }
  cvt(wd, bfp(off_wdb), (size_t)INTERM * HIDN);
  gemm_bf16_kernel<<<dim3(SEQ / 64, HIDN / 128), 256, GEMM_LDS, stream>>>(
      bfp(off_hb), bfp(off_wdb), ffp(off_mlpf), SEQ, HIDN, INTERM);
  add_rmsnorm_kernel<<<SEQ, 256, 0, stream>>>(ffp(off_x1f), ffp(off_mlpf),
                                              (float*)d_out, nullptr);
}
